// Mamba2Block_81587198755156
// MI455X (gfx1250) — matrix operand order repaired and checked by reading, not compiled, not run on hardware
//
#include <hip/hip_runtime.h>
#include <hip/hip_bf16.h>
#include <math.h>

#define BATCH   2
#define SEQLEN  1024
#define DMODEL  1024
#define DSTATE  64
#define DCONV   4
#define HEADDIM 64
#define DINNER  2048
#define NHEADS  32
#define CONVDIM 2176
#define DINPROJ 4256
#define NPAD1   4352          // DINPROJ padded to multiple of 256
#define MTOTAL  (BATCH * SEQLEN)
#define RMS_EPS 1.1920929e-07f

typedef __attribute__((ext_vector_type(16))) __bf16 bf16x16;
typedef __attribute__((ext_vector_type(8)))  float  f32x8;

union BFrag { unsigned short u[16]; bf16x16 v; };

__device__ __forceinline__ unsigned short f32_to_bf16(float f) {
  unsigned int u = __float_as_uint(f);
  u += 0x7FFFu + ((u >> 16) & 1u);           // round-to-nearest-even
  return (unsigned short)(u >> 16);
}

// Async DMA: 16 bytes global -> LDS, tracked by ASYNCcnt (per-lane transfer).
// Generic pointers to LDS carry the wave-relative LDS offset in the low 32 bits.
__device__ __forceinline__ void async_copy_b128(void* lds_ptr, const void* gptr) {
  const unsigned lds_off = (unsigned)(unsigned long long)(uintptr_t)lds_ptr;
  asm volatile("global_load_async_to_lds_b128 %0, %1, off"
               :: "v"(lds_off), "v"((unsigned long long)(uintptr_t)gptr)
               : "memory");
}

// ---------------------------------------------------------------------------
// bf16 GEMM with async-LDS double buffering.
// C[M,N] = A[M,K] (bf16) * Bt[K,ldb] (bf16, already transposed/padded).
// Block: 256 thr, tile 32(M) x 256(N); wave tile 16x64 (4 WMMA accumulators).
// ---------------------------------------------------------------------------
__global__ __launch_bounds__(256) void gemm_bf16_wmma(
    const unsigned short* __restrict__ A,
    const unsigned short* __restrict__ Bt,
    float* __restrict__ C,
    int M, int N, int K, int ldb)
{
  __shared__ __align__(16) unsigned short As[2][32][40];    // [buf][m][k]
  __shared__ __align__(16) unsigned short Bs[2][32][264];   // [buf][k][n]

  const int tid  = threadIdx.x;
  const int lane = tid & 31;
  const int w    = tid >> 5;       // 0..7
  const int wm   = w >> 2;         // 0..1
  const int wn   = w & 3;          // 0..3
  const int m0   = blockIdx.y * 32;
  const int n0   = blockIdx.x * 256;

  f32x8 acc[4] = {};

  // A stage: 32x32 halves = 128 x 16B segments; even lanes issue (1 op/wave).
  const int aseg    = tid >> 1;           // 0..127
  const int arow    = aseg >> 2;          // 0..31
  const int acol    = (aseg & 3) * 8;     // halves
  const bool a_lane = (tid & 1) == 0;

  const int niter = K / 32;

  auto issue = [&](int buf, int kb) {
    if (a_lane)
      async_copy_b128(&As[buf][arow][acol],
                      A + (size_t)(m0 + arow) * K + kb + acol);
    #pragma unroll
    for (int q = 0; q < 4; ++q) {           // 4 ops/wave
      const int seg  = tid + q * 256;       // 0..1023
      const int krow = seg >> 5;            // 0..31
      const int scol = (seg & 31) * 8;      // halves
      async_copy_b128(&Bs[buf][krow][scol],
                      Bt + (size_t)(kb + krow) * ldb + n0 + scol);
    }
  };

  issue(0, 0);

  for (int i = 0; i < niter; ++i) {
    const int cur = i & 1;
    if (i + 1 < niter) {
      issue(cur ^ 1, (i + 1) * 32);
      // 5 async issues/wave/tile; in-order completion => oldest tile resident
      asm volatile("s_wait_asynccnt 0x5" ::: "memory");
    } else {
      asm volatile("s_wait_asynccnt 0x0" ::: "memory");
    }
    __syncthreads();

    // A fragment (ISA 16-bit A 16x32 layout): lane&15 = M row
    BFrag af;
    {
      const int row   = wm * 16 + (lane & 15);
      const int kbase = (lane >> 4) * 8;
      #pragma unroll
      for (int j = 0; j < 16; ++j)
        af.u[j] = As[cur][row][((j & 8) ? 16 : 0) + kbase + (j & 7)];
    }
    // B fragments: lane&15 = N column, K in same table order as A
    #pragma unroll
    for (int s = 0; s < 4; ++s) {
      BFrag bf;
      const int nb = wn * 64 + s * 16;
      #pragma unroll
      for (int j = 0; j < 16; ++j) bf.u[j] = Bs[cur][((j & 8) ? 16 : 0) + (lane >> 4) * 8 + (j & 7)][nb + (lane & 15)];
      acc[s] = __builtin_amdgcn_wmma_f32_16x16x32_bf16(
          false, af.v, false, bf.v, (short)0, acc[s], false, false);
    }
    __syncthreads();
  }

  // C/D layout: VGPR r -> M = r + 8*(lane>=16); N = lane&15
  const int row0 = m0 + wm * 16 + ((lane >> 4) * 8);
  const int coln = n0 + wn * 64 + (lane & 15);
  #pragma unroll
  for (int s = 0; s < 4; ++s) {
    const int col = coln + s * 16;
    if (col < N) {
      #pragma unroll
      for (int r = 0; r < 8; ++r)
        C[(size_t)(row0 + r) * N + col] = acc[s][r];
    }
  }
}

// ---------------------------------------------------------------------------
// One-time operand preparation
// ---------------------------------------------------------------------------
__global__ void convert_bf16_kernel(const float* __restrict__ src,
                                    unsigned short* __restrict__ dst, int n) {
  const int i = blockIdx.x * blockDim.x + threadIdx.x;
  if (i < n) dst[i] = f32_to_bf16(src[i]);
}

// W[N,K] f32 -> Wt[K,Npad] bf16 (zero-padded columns), LDS-tiled transpose
__global__ __launch_bounds__(256) void transpose_to_bf16(
    const float* __restrict__ W, unsigned short* __restrict__ Wt,
    int N, int K, int Npad)
{
  __shared__ float tile[32][33];
  const int tx = threadIdx.x;        // 0..31
  const int ty = threadIdx.y;        // 0..7
  const int nb = blockIdx.x * 32;
  const int kb = blockIdx.y * 32;
  #pragma unroll
  for (int r = 0; r < 32; r += 8) {
    const int n = nb + ty + r;
    tile[ty + r][tx] = (n < N) ? W[(size_t)n * K + kb + tx] : 0.0f;
  }
  __syncthreads();
  #pragma unroll
  for (int r = 0; r < 32; r += 8) {
    const int k = kb + ty + r;
    Wt[(size_t)k * Npad + nb + tx] = f32_to_bf16(tile[tx][ty + r]);
  }
}

// ---------------------------------------------------------------------------
// Elementwise / scan stages
// ---------------------------------------------------------------------------
__global__ void conv_silu_kernel(const float* __restrict__ zxbcdt,
                                 const float* __restrict__ conv_w,
                                 const float* __restrict__ conv_b,
                                 float* __restrict__ xBCc)
{
  int idx = blockIdx.x * blockDim.x + threadIdx.x;
  if (idx >= MTOTAL * CONVDIM) return;
  const int c   = idx % CONVDIM;
  const int row = idx / CONVDIM;
  const int t   = row % SEQLEN;
  const int b   = row / SEQLEN;
  float acc = conv_b[c];
  #pragma unroll
  for (int k = 0; k < DCONV; ++k) {
    const int l = t - (DCONV - 1) + k;
    if (l >= 0)
      acc += zxbcdt[(size_t)(b * SEQLEN + l) * DINPROJ + DINNER + c] *
             conv_w[c * DCONV + k];
  }
  xBCc[idx] = acc / (1.0f + expf(-acc));   // SiLU
}

__global__ void dt_softplus_kernel(const float* __restrict__ zxbcdt,
                                   const float* __restrict__ dt_bias,
                                   float* __restrict__ dt_out)
{
  int idx = blockIdx.x * blockDim.x + threadIdx.x;
  if (idx >= MTOTAL * NHEADS) return;
  const int h   = idx % NHEADS;
  const int row = idx / NHEADS;
  const float v = zxbcdt[(size_t)row * DINPROJ + DINNER + CONVDIM + h] + dt_bias[h];
  dt_out[idx] = (v > 20.0f) ? v : log1pf(expf(v));
}

// Sequential selective scan: 4 heads per 256-thread block; thread = (head,p);
// 64-element state row in VGPRs; B/C broadcast via LDS (shared, NGROUPS=1).
__global__ __launch_bounds__(256) void ssm_scan_kernel(
    const float* __restrict__ xBCc, const float* __restrict__ dt,
    const float* __restrict__ A_log, const float* __restrict__ D_param,
    float* __restrict__ y)
{
  __shared__ float BC[128];
  const int tid = threadIdx.x;
  const int p   = tid & 63;
  const int hh  = tid >> 6;
  const int b   = blockIdx.x >> 3;
  const int h   = (blockIdx.x & 7) * 4 + hh;
  const float Ah = -expf(A_log[h]);
  const float Dh = D_param[h];

  float hstate[DSTATE];
  #pragma unroll
  for (int n = 0; n < DSTATE; ++n) hstate[n] = 0.0f;

  for (int t = 0; t < SEQLEN; ++t) {
    const size_t row = (size_t)b * SEQLEN + t;
    if (tid < 128) BC[tid] = xBCc[row * CONVDIM + DINNER + tid];
    __syncthreads();

    const float dtv  = dt[row * NHEADS + h];
    const float dA   = expf(dtv * Ah);
    const float xv   = xBCc[row * CONVDIM + h * HEADDIM + p];
    const float coef = dtv * xv;
    float acc = 0.0f;
    #pragma unroll
    for (int n = 0; n < DSTATE; ++n) {
      hstate[n] = fmaf(hstate[n], dA, coef * BC[n]);
      acc = fmaf(hstate[n], BC[64 + n], acc);
    }
    y[row * DINNER + h * HEADDIM + p] = acc + Dh * xv;
    __syncthreads();
  }
}

// RMSNorm + SiLU(z) gate; emits bf16 directly (GEMM2's A operand)
__global__ __launch_bounds__(256) void rmsnorm_gate_kernel(
    const float* __restrict__ y, const float* __restrict__ zxbcdt,
    const float* __restrict__ norm_w, unsigned short* __restrict__ ygb)
{
  __shared__ float red[256];
  const int row = blockIdx.x;
  const int tid = threadIdx.x;
  float s = 0.0f;
  #pragma unroll
  for (int i = 0; i < DINNER / 256; ++i) {
    const float v = y[(size_t)row * DINNER + tid + i * 256];
    s += v * v;
  }
  red[tid] = s;
  __syncthreads();
  for (int off = 128; off > 0; off >>= 1) {
    if (tid < off) red[tid] += red[tid + off];
    __syncthreads();
  }
  const float scale = rsqrtf(red[0] / (float)DINNER + RMS_EPS);
  #pragma unroll
  for (int i = 0; i < DINNER / 256; ++i) {
    const int d = tid + i * 256;
    float v = y[(size_t)row * DINNER + d] * scale * norm_w[d];
    const float z = zxbcdt[(size_t)row * DINPROJ + d];
    v *= z / (1.0f + expf(-z));
    ygb[(size_t)row * DINNER + d] = f32_to_bf16(v);
  }
}

// ---------------------------------------------------------------------------
extern "C" void kernel_launch(void* const* d_in, const int* in_sizes, int n_in,
                              void* d_out, int out_size, void* d_ws, size_t ws_size,
                              hipStream_t stream) {
  (void)in_sizes; (void)n_in; (void)out_size; (void)ws_size;
  const float* x       = (const float*)d_in[0];
  const float* W_in    = (const float*)d_in[1];
  const float* conv_w  = (const float*)d_in[2];
  const float* conv_b  = (const float*)d_in[3];
  const float* dt_bias = (const float*)d_in[4];
  const float* A_log   = (const float*)d_in[5];
  const float* D_param = (const float*)d_in[6];
  const float* norm_w  = (const float*)d_in[7];
  const float* W_out   = (const float*)d_in[8];
  float* out = (float*)d_out;

  // workspace layout (float units; every section is a multiple of 4 floats)
  float* ws = (float*)d_ws;
  size_t off = 0;
  float* zxbcdt = ws + off; off += (size_t)MTOTAL * DINPROJ;     // f32
  float* xBCc   = ws + off; off += (size_t)MTOTAL * CONVDIM;     // f32
  float* dtw    = ws + off; off += (size_t)MTOTAL * NHEADS;      // f32
  float* yw     = ws + off; off += (size_t)MTOTAL * DINNER;      // f32
  unsigned short* xb    = (unsigned short*)(ws + off); off += (size_t)MTOTAL * DMODEL / 2;
  unsigned short* Wint  = (unsigned short*)(ws + off); off += (size_t)DMODEL * NPAD1 / 2;
  unsigned short* Woutt = (unsigned short*)(ws + off); off += (size_t)DINNER * DMODEL / 2;
  unsigned short* ygb   = (unsigned short*)(ws + off); off += (size_t)MTOTAL * DINNER / 2;

  // 0) one-time bf16 conversion / weight transposes
  convert_bf16_kernel<<<(MTOTAL * DMODEL + 255) / 256, 256, 0, stream>>>(
      x, xb, MTOTAL * DMODEL);
  transpose_to_bf16<<<dim3(NPAD1 / 32, DMODEL / 32), dim3(32, 8), 0, stream>>>(
      W_in, Wint, DINPROJ, DMODEL, NPAD1);
  transpose_to_bf16<<<dim3(DMODEL / 32, DINNER / 32), dim3(32, 8), 0, stream>>>(
      W_out, Woutt, DMODEL, DINNER, DMODEL);

  // 1) zxbcdt = x @ W_in^T
  gemm_bf16_wmma<<<dim3(NPAD1 / 256, MTOTAL / 32), 256, 0, stream>>>(
      xb, Wint, zxbcdt, MTOTAL, DINPROJ, DMODEL, NPAD1);

  // 2) conv + SiLU, 3) dt softplus
  conv_silu_kernel<<<(MTOTAL * CONVDIM + 255) / 256, 256, 0, stream>>>(
      zxbcdt, conv_w, conv_b, xBCc);
  dt_softplus_kernel<<<(MTOTAL * NHEADS + 255) / 256, 256, 0, stream>>>(
      zxbcdt, dt_bias, dtw);

  // 4) selective scan
  ssm_scan_kernel<<<BATCH * (NHEADS / 4), 256, 0, stream>>>(
      xBCc, dtw, A_log, D_param, yw);

  // 5) RMSNorm + gate -> bf16
  rmsnorm_gate_kernel<<<MTOTAL, 256, 0, stream>>>(yw, zxbcdt, norm_w, ygb);

  // 6) out = y_gated @ W_out^T
  gemm_bf16_wmma<<<dim3(DMODEL / 256, MTOTAL / 32), 256, 0, stream>>>(
      ygb, Woutt, out, MTOTAL, DMODEL, DINNER, DMODEL);
}
